// MultimodalSAttention_7215545057878
// MI455X (gfx1250) — compile-verified, matrix operand order checked by reading, not run on hardware
//
#include <hip/hip_runtime.h>
#include <hip/hip_bf16.h>
#include <math.h>

// Problem constants from the reference (fixed by setup_inputs).
static constexpr int B_   = 8;     // batch
static constexpr int Ntok = 8192;  // tokens per batch
static constexpr int Lk   = 512;   // feature dim of x (K for score GEMM, N for pooling GEMM)
static constexpr int Dh   = 128;   // MLP hidden dim
static constexpr int Mq   = 16;    // max_queries / images per batch

typedef __attribute__((ext_vector_type(16))) __bf16          bf16x16;
typedef __attribute__((ext_vector_type(16))) unsigned short  u16x16;
typedef __attribute__((ext_vector_type(8)))  float           f32x8;

// fp32 -> bf16, round-to-nearest-even (bit-exact, no reliance on __bf16 arithmetic)
__device__ __forceinline__ unsigned short f2bf(float f) {
  union { float f; unsigned u; } v; v.f = f;
  unsigned r = v.u + 0x7FFFu + ((v.u >> 16) & 1u);
  return (unsigned short)(r >> 16);
}

// ---------------------------------------------------------------------------
// Kernel 0: W1 [L,D] fp32  ->  W1T [D,L] bf16  (so B-matrix K-runs are contiguous)
// ---------------------------------------------------------------------------
__global__ void prep_w1t(const float* __restrict__ W1, unsigned short* __restrict__ w1t) {
  int idx = blockIdx.x * blockDim.x + threadIdx.x;
  if (idx >= Lk * Dh) return;
  int k = idx / Dh, d = idx % Dh;
  w1t[d * Lk + k] = f2bf(W1[idx]);
}

// ---------------------------------------------------------------------------
// Kernel 1: scores[t] = tanh(x[t,:] @ W1) @ W2 for a 16-token tile per wave.
// M=16 tokens, N=128 (8 tiles of 16), K=512 (16 steps of 32) -> 128 WMMAs/wave.
// ---------------------------------------------------------------------------
__global__ void __launch_bounds__(32)
scores_kernel(const float* __restrict__ x,
              const unsigned short* __restrict__ w1t,
              const float* __restrict__ W2,
              float* __restrict__ sc) {
  const int  lane = threadIdx.x;      // 0..31, full wave active (WMMA needs EXEC all-ones)
  const int  half = lane >> 4;        // 0 or 1
  const int  row  = lane & 15;        // A-matrix row M (and B-matrix column within tile)
  const long t0   = (long)blockIdx.x * 16;              // flat token tile base over b*n
  const float* xr = x + (t0 + row) * (long)Lk;

  f32x8 acc[8] = {};

  for (int k0 = 0; k0 < Lk; k0 += 32) {
    // ---- A tile: x[16 tok x 32 K] fp32 -> bf16, per 16-bit A layout:
    // lane(0-15): K = {0..7, 16..23}; lane(16-31): K = {8..15, 24..31}
    u16x16 au;
    const float* p0 = xr + k0 + half * 8;
    #pragma unroll
    for (int j = 0; j < 8; ++j) au[j]     = f2bf(p0[j]);
    #pragma unroll
    for (int j = 0; j < 8; ++j) au[8 + j] = f2bf(p0[16 + j]);
    const bf16x16 a = __builtin_bit_cast(bf16x16, au);

    // ---- 8 N-tiles of W1 (columns of D), contiguous bf16 runs from W1T
    #pragma unroll
    for (int nt = 0; nt < 8; ++nt) {
      const int col = nt * 16 + row;
      const unsigned short* bp = w1t + (size_t)col * Lk + k0 + half * 8;
      union { uint4 v; unsigned short s[8]; } c0, c1;
      c0.v = *(const uint4*)(bp);        // K = half*8 + 0..7
      c1.v = *(const uint4*)(bp + 16);   // K = 16 + half*8 + 0..7
      u16x16 bu;
      #pragma unroll
      for (int j = 0; j < 8; ++j) { bu[j] = c0.s[j]; bu[8 + j] = c1.s[j]; }
      const bf16x16 b = __builtin_bit_cast(bf16x16, bu);
      acc[nt] = __builtin_amdgcn_wmma_f32_16x16x32_bf16(
          false, a, false, b, (short)0, acc[nt], false, false);
    }
  }

  // Epilogue: score[M] = sum_d tanh(h[M,d]) * W2[d].
  // C layout: lane(0-15) col=lane holds M=v (v=0..7); lane(16-31) holds M=v+8.
  float pp[8];
  #pragma unroll
  for (int v = 0; v < 8; ++v) pp[v] = 0.f;
  #pragma unroll
  for (int nt = 0; nt < 8; ++nt) {
    const float w2v = W2[nt * 16 + row];
    #pragma unroll
    for (int v = 0; v < 8; ++v) pp[v] += tanhf(acc[nt][v]) * w2v;
  }
  // Reduce across the 16 columns within each half-wave.
  #pragma unroll
  for (int v = 0; v < 8; ++v) {
    #pragma unroll
    for (int off = 8; off >= 1; off >>= 1) pp[v] += __shfl_xor(pp[v], off, 32);
  }
  if (row == 0) {            // lanes 0 (rows 0..7) and 16 (rows 8..15)
    #pragma unroll
    for (int v = 0; v < 8; ++v) sc[t0 + half * 8 + v] = pp[v];
  }
}

// ---------------------------------------------------------------------------
// Kernel 2: per-(b,image) masked softmax stats (max, 1/sum-exp), online + LDS tree.
// ---------------------------------------------------------------------------
__global__ void __launch_bounds__(256)
stats_kernel(const float* __restrict__ sc,
             const int* __restrict__ ids,
             const unsigned char* __restrict__ pad,   // bool mask: nonzero = padding
             float* __restrict__ stats) {
  const int bm = blockIdx.x;            // 0..B*Mq-1
  const int bb = bm / Mq, m = bm % Mq;
  const int tid = threadIdx.x;

  float mx = -INFINITY, sm = 0.f;
  for (int j = tid; j < Ntok; j += 256) {
    const int tok = bb * Ntok + j;
    if (ids[tok] == m && pad[tok] == 0) {
      const float s = sc[tok];
      if (s > mx) { sm = sm * __expf(mx - s) + 1.f; mx = s; }
      else        { sm += __expf(s - mx); }
    }
  }

  __shared__ float smx[256], ssm[256];
  smx[tid] = mx; ssm[tid] = sm;
  __syncthreads();
  for (int st = 128; st >= 1; st >>= 1) {
    if (tid < st) {
      float mx1 = smx[tid], sm1 = ssm[tid];
      float mx2 = smx[tid + st], sm2 = ssm[tid + st];
      if (mx2 > mx1) { float t = mx1; mx1 = mx2; mx2 = t; t = sm1; sm1 = sm2; sm2 = t; }
      if (sm2 > 0.f) sm1 += sm2 * __expf(mx2 - mx1);
      smx[tid] = mx1; ssm[tid] = sm1;
    }
    __syncthreads();
  }
  if (tid == 0) {
    const float s0 = ssm[0];
    stats[bm * 2 + 0] = smx[0];
    stats[bm * 2 + 1] = (s0 > 0.f) ? 1.f / s0 : 0.f;
  }
}

// ---------------------------------------------------------------------------
// Kernel 3: out[b,m,L] = softmax-weights[m,n] @ x[n,L].
// One wave per (b, 16-wide L slice). K = 8192 tokens in 32-token steps; weights
// computed on the fly from scores+stats; x tile staged fp32->bf16 through LDS.
// ---------------------------------------------------------------------------
__global__ void __launch_bounds__(32)
pool_kernel(const float* __restrict__ x,
            const float* __restrict__ sc,
            const int* __restrict__ ids,
            const unsigned char* __restrict__ pad,
            const float* __restrict__ stats,
            float* __restrict__ out) {
  const int lane = threadIdx.x;
  const int half = lane >> 4;
  const int nl   = lane & 15;                 // A-row m, and B/C column
  const int bb   = blockIdx.x / (Lk / 16);
  const int lt   = blockIdx.x % (Lk / 16);

  const float mx_m  = stats[(bb * Mq + nl) * 2 + 0];
  const float inv_m = stats[(bb * Mq + nl) * 2 + 1];

  __shared__ float          s_sc[32];
  __shared__ int            s_id[32];
  __shared__ unsigned short s_x[32 * 16];     // 32 tokens x 16 cols, bf16

  f32x8 acc = {};

  for (int k0 = 0; k0 < Ntok; k0 += 32) {
    const int tok = bb * Ntok + k0 + lane;
    s_sc[lane] = sc[tok];
    s_id[lane] = pad[tok] ? -1 : ids[tok];
    const float* xr = x + (size_t)tok * Lk + lt * 16;   // coalesced 64B per lane
    #pragma unroll
    for (int j = 0; j < 16; ++j) s_x[lane * 16 + j] = f2bf(xr[j]);
    __syncthreads();

    u16x16 au, bu;
    #pragma unroll
    for (int e = 0; e < 16; ++e) {
      // local K index within the 32-token chunk, per 16-bit A/B lane layout
      const int kk = (e < 8) ? (half * 8 + e) : (16 + half * 8 + (e - 8));
      const float w = (s_id[kk] == nl) ? __expf(s_sc[kk] - mx_m) * inv_m : 0.f;
      au[e] = f2bf(w);
      bu[e] = s_x[kk * 16 + nl];
    }
    const bf16x16 a = __builtin_bit_cast(bf16x16, au);
    const bf16x16 b = __builtin_bit_cast(bf16x16, bu);
    acc = __builtin_amdgcn_wmma_f32_16x16x32_bf16(
        false, a, false, b, (short)0, acc, false, false);
    __syncthreads();
  }

  // C layout: lane(0-15) -> M = v, lane(16-31) -> M = v + 8; col = lt*16 + nl.
  const int col = lt * 16 + nl;
  #pragma unroll
  for (int v = 0; v < 8; ++v) {
    const int mrow = v + half * 8;
    out[(size_t)(bb * Mq + mrow) * Lk + col] = acc[v];
  }
}

// ---------------------------------------------------------------------------
extern "C" void kernel_launch(void* const* d_in, const int* in_sizes, int n_in,
                              void* d_out, int out_size, void* d_ws, size_t ws_size,
                              hipStream_t stream) {
  const float*         x   = (const float*)d_in[0];           // [b,n,L] fp32
  const float*         W1  = (const float*)d_in[1];           // [L,D]   fp32
  const float*         W2  = (const float*)d_in[2];           // [D,1]   fp32
  const int*           ids = (const int*)d_in[3];             // [b,n]   int32
  const unsigned char* pad = (const unsigned char*)d_in[4];   // [b,n]   bool (1 byte)
  // d_in[5] = max_queries scalar (== Mq, compile-time here)

  unsigned char*  ws    = (unsigned char*)d_ws;
  unsigned short* w1t   = (unsigned short*)(ws);                       // 128 KB bf16 [D,L]
  float*          sc    = (float*)(ws + 131072);                       // 256 KB scores
  float*          stats = (float*)(ws + 131072 + 262144);              //   1 KB (max, 1/sum)
  float*          out   = (float*)d_out;                               // [b*Mq, L] fp32

  prep_w1t     <<<(Lk * Dh + 255) / 256, 256, 0, stream>>>(W1, w1t);
  scores_kernel<<<(B_ * Ntok) / 16,       32, 0, stream>>>(x, w1t, W2, sc);
  stats_kernel <<<B_ * Mq,               256, 0, stream>>>(sc, ids, pad, stats);
  pool_kernel  <<<B_ * (Lk / 16),         32, 0, stream>>>(x, sc, ids, pad, stats, out);
}
